// attention_module_17308718203307
// MI455X (gfx1250) — compile-verified, matrix operand order checked by reading, not run on hardware
//
#include <hip/hip_runtime.h>
#include <hip/hip_bf16.h>
#include <math.h>

typedef __attribute__((ext_vector_type(16))) _Float16 v16h;
typedef __attribute__((ext_vector_type(8)))  float    v8f;

#define DIM1 256
#define DIM2 128
#define NSEG 512

// ---------- float <-> order-preserving uint encoding (for atomic max) ----------
__device__ __forceinline__ unsigned enc_f32(float f) {
    unsigned u = __float_as_uint(f);
    return (u & 0x80000000u) ? ~u : (u | 0x80000000u);
}
__device__ __forceinline__ float dec_f32(unsigned e) {
    unsigned u = (e & 0x80000000u) ? (e ^ 0x80000000u) : ~e;
    return __uint_as_float(u);
}

// ---------- 0: zero output + softmax stats ----------
__global__ void init_kernel(float* __restrict__ out, int out_n,
                            unsigned* __restrict__ smax, float* __restrict__ ssum) {
    int i = blockIdx.x * 256 + threadIdx.x;
    if (i < out_n)                 out[i] = 0.0f;
    else if (i < out_n + NSEG)     smax[i - out_n] = 0u;          // enc(-inf) > 0, so 0 is identity
    else if (i < out_n + 2*NSEG)   ssum[i - out_n - NSEG] = 0.0f;
}

// ---------- 1: pre-swizzle Wq/Wg (f32) into per-lane f16 WMMA B-fragments ----------
// Layout: wfrag[((trans*64 + jt*8 + ks)*32 + lane)*16 + t]
// B(k, j) = W[j, k]; lane (r,h) holds column j = jt*16 + r,
// halves t=0..7  -> K = 32*ks + 8*h + t
// halves t=8..15 -> K = 32*ks + 16 + 8*h + (t-8)   (mirrors 16-bit A-matrix layout)
__global__ void prep_weights_kernel(const float* __restrict__ wq,
                                    const float* __restrict__ wg,
                                    _Float16* __restrict__ wfrag) {
    int tid = blockIdx.x * 256 + threadIdx.x;      // 0 .. 4095
    if (tid >= 2 * 64 * 32) return;
    int lane = tid & 31;
    int fidx = tid >> 5;                           // trans*64 + jt*8 + ks
    int trans = fidx >> 6;
    int jt = (fidx >> 3) & 7;
    int ks = fidx & 7;
    const float* W = trans ? wg : wq;
    int r = lane & 15, h = lane >> 4;
    int j = jt * 16 + r;
    _Float16* dst = wfrag + (size_t)tid * 16;
    #pragma unroll
    for (int t = 0; t < 16; ++t) {
        int K = ks * 32 + ((t >> 3) << 4) + h * 8 + (t & 7);
        dst[t] = (_Float16)W[j * DIM1 + K];
    }
}

// ---------- 2: fused WMMA GEMM x Wq^T / x Wg^T + tanh*sigmoid*w -> key[N] ----------
__global__ __launch_bounds__(128) void gemm_key_kernel(
    const float* __restrict__ x,
    const _Float16* __restrict__ wfrag,
    const float* __restrict__ weight,
    float* __restrict__ key, int nTiles) {

    int wave = threadIdx.x >> 5;
    int lane = threadIdx.x & 31;
    int tile = blockIdx.x * 4 + wave;
    if (tile >= nTiles) return;                    // uniform per wave: EXEC stays all-ones

    int r = lane & 15;                             // A-row / B-col / C-col index
    int h = lane >> 4;                             // K-half selector (A/B), M-half (C/D)
    const float* xrow = x + (size_t)(tile * 16 + r) * DIM1;

    // Load 8 A-fragments (K = 0..255) for this 16-row tile, f32 -> f16
    v16h afrag[8];
    #pragma unroll
    for (int ks = 0; ks < 8; ++ks) {
        int kb = ks * 32 + h * 8;
        float tmp[16];
        *(float4*)(tmp + 0)  = *(const float4*)(xrow + kb + 0);
        *(float4*)(tmp + 4)  = *(const float4*)(xrow + kb + 4);
        *(float4*)(tmp + 8)  = *(const float4*)(xrow + kb + 16);
        *(float4*)(tmp + 12) = *(const float4*)(xrow + kb + 20);
        #pragma unroll
        for (int t = 0; t < 16; ++t) afrag[ks][t] = (_Float16)tmp[t];
    }

    float keyp[8];
    #pragma unroll
    for (int v = 0; v < 8; ++v) keyp[v] = 0.0f;

    const v16h* fq = (const v16h*)wfrag;           // fragment stride over ks = 32 v16h
    #pragma unroll
    for (int jt = 0; jt < 8; ++jt) {
        v8f accq = {};
        v8f accg = {};
        const v16h* bq = fq + ((0 * 64 + jt * 8) * 32 + lane);
        const v16h* bg = fq + ((1 * 64 + jt * 8) * 32 + lane);
        #pragma unroll
        for (int ks = 0; ks < 8; ++ks) {
            v16h bqf = bq[ks * 32];
            v16h bgf = bg[ks * 32];
            accq = __builtin_amdgcn_wmma_f32_16x16x32_f16(
                       false, afrag[ks], false, bqf, (short)0, accq, false, false);
            accg = __builtin_amdgcn_wmma_f32_16x16x32_f16(
                       false, afrag[ks], false, bgf, (short)0, accg, false, false);
        }
        // C/D layout: acc[v] is element (row = h*8+v, col = jt*16 + r)
        float wv = weight[jt * 16 + r];
        #pragma unroll
        for (int v = 0; v < 8; ++v) {
            float q = tanhf(accq[v]);
            float g = 1.0f / (1.0f + __expf(-accg[v]));
            keyp[v] += q * g * wv;
        }
    }

    // Reduce over the 16 column-lanes sharing the same h
    #pragma unroll
    for (int m = 1; m < 16; m <<= 1) {
        #pragma unroll
        for (int v = 0; v < 8; ++v)
            keyp[v] += __shfl_xor(keyp[v], m, 16);
    }
    if (r == 0) {
        #pragma unroll
        for (int v = 0; v < 8; ++v)
            key[(size_t)tile * 16 + h * 8 + v] = keyp[v];
    }
}

// ---------- 3: segment max (LDS-privatized, then global atomic) ----------
#define SEG_CHUNK 4096
__global__ __launch_bounds__(256) void seg_max_kernel(
    const float* __restrict__ key, const long long* __restrict__ batch,
    unsigned* __restrict__ smax_g, int n) {
    __shared__ unsigned smax[NSEG];
    int tid = threadIdx.x;
    for (int g = tid; g < NSEG; g += 256) smax[g] = 0u;
    __syncthreads();
    int start = blockIdx.x * SEG_CHUNK;
    int end = min(start + SEG_CHUNK, n);
    for (int i = start + tid; i < end; i += 256) {
        int b = (int)batch[i];
        atomicMax(&smax[b], enc_f32(key[i]));
    }
    __syncthreads();
    for (int g = tid; g < NSEG; g += 256)
        if (smax[g] != 0u) atomicMax(&smax_g[g], smax[g]);
}

// ---------- 4: segment sum of exp(key - max) ----------
__global__ __launch_bounds__(256) void seg_sum_kernel(
    const float* __restrict__ key, const long long* __restrict__ batch,
    const unsigned* __restrict__ smax_g, float* __restrict__ ssum_g, int n) {
    __shared__ float ssum[NSEG];
    int tid = threadIdx.x;
    for (int g = tid; g < NSEG; g += 256) ssum[g] = 0.0f;
    __syncthreads();
    int start = blockIdx.x * SEG_CHUNK;
    int end = min(start + SEG_CHUNK, n);
    for (int i = start + tid; i < end; i += 256) {
        int b = (int)batch[i];
        float mx = dec_f32(smax_g[b]);
        atomicAdd(&ssum[b], __expf(key[i] - mx));
    }
    __syncthreads();
    for (int g = tid; g < NSEG; g += 256)
        if (ssum[g] != 0.0f) atomicAdd(&ssum_g[g], ssum[g]);
}

// ---------- 5: attention-weighted pooling (one coalesced pass over x) ----------
#define POOL_CHUNK 256
__global__ __launch_bounds__(256) void pool_kernel(
    const float* __restrict__ x, const long long* __restrict__ batch,
    const float* __restrict__ key, const unsigned* __restrict__ smax_g,
    const float* __restrict__ ssum_g, float* __restrict__ out, int n) {
    __shared__ float s_att[POOL_CHUNK];
    __shared__ int   s_b[POOL_CHUNK];
    int tid = threadIdx.x;
    int c0 = blockIdx.x * POOL_CHUNK;
    int cnt = min(POOL_CHUNK, n - c0);
    if (tid < cnt) {
        int i = c0 + tid;
        int b = (int)batch[i];
        float mx = dec_f32(smax_g[b]);
        s_att[tid] = __expf(key[i] - mx) / ssum_g[b];
        s_b[tid] = b;
    }
    __syncthreads();
    int d = tid;                                   // feature index, fully coalesced over x
    float acc = 0.0f;
    int cur = s_b[0];
    for (int k = 0; k < cnt; ++k) {
        int b = s_b[k];
        if (b != cur) {                            // sorted batch -> rare flush
            atomicAdd(&out[(size_t)cur * DIM1 + d], acc);
            acc = 0.0f; cur = b;
        }
        acc += x[(size_t)(c0 + k) * DIM1 + d] * s_att[k];
    }
    atomicAdd(&out[(size_t)cur * DIM1 + d], acc);
}

extern "C" void kernel_launch(void* const* d_in, const int* in_sizes, int n_in,
                              void* d_out, int out_size, void* d_ws, size_t ws_size,
                              hipStream_t stream) {
    const float*     x      = (const float*)d_in[0];
    const long long* batch  = (const long long*)d_in[1];
    const float*     wq     = (const float*)d_in[2];
    const float*     wg     = (const float*)d_in[3];
    const float*     weight = (const float*)d_in[4];
    float*           out    = (float*)d_out;

    const int n = in_sizes[1];                     // N (number of nodes)
    const int nTiles = (n + 15) / 16;

    // workspace layout
    char* ws = (char*)d_ws;
    _Float16* wfrag = (_Float16*)ws;               // 2*64*32*16 halves = 128 KB
    size_t off = (size_t)2 * 64 * 32 * 16 * sizeof(_Float16);
    float* key = (float*)(ws + off);               // N floats
    off += (size_t)nTiles * 16 * sizeof(float);
    unsigned* smax = (unsigned*)(ws + off);        // 512 uints
    off += NSEG * sizeof(unsigned);
    float* ssum = (float*)(ws + off);              // 512 floats

    int initN = out_size + 2 * NSEG;
    init_kernel<<<(initN + 255) / 256, 256, 0, stream>>>(out, out_size, smax, ssum);
    prep_weights_kernel<<<(2 * 64 * 32 + 255) / 256, 256, 0, stream>>>(wq, wg, wfrag);
    gemm_key_kernel<<<(nTiles + 3) / 4, 128, 0, stream>>>(x, wfrag, weight, key, nTiles);
    int segBlocks = (n + SEG_CHUNK - 1) / SEG_CHUNK;
    seg_max_kernel<<<segBlocks, 256, 0, stream>>>(key, batch, smax, n);
    seg_sum_kernel<<<segBlocks, 256, 0, stream>>>(key, batch, smax, ssum, n);
    pool_kernel<<<(n + POOL_CHUNK - 1) / POOL_CHUNK, 256, 0, stream>>>(
        x, batch, key, smax, ssum, out, n);
}
